// ParametericDRSolver_67851893342490
// MI455X (gfx1250) — matrix operand order repaired and checked by reading, not compiled, not run on hardware
//
#include <hip/hip_runtime.h>

// Problem constants (from reference)
#define KB    16384   // batch
#define KPARM 128
#define KHID  960
#define KN    96      // N = X_DIM + N_INEQ
#define KM    64      // M = N_EQ + N_INEQ
#define KLBC  1e-8f
#define KUBC  1e5f

typedef __attribute__((ext_vector_type(16))) _Float16 v16h;
typedef __attribute__((ext_vector_type(8)))  float    v8f;

__device__ inline float wred(float v) {
#pragma unroll
  for (int o = 16; o > 0; o >>= 1) v += __shfl_xor(v, o);
  return v;
}

// ---------------------------------------------------------------------------
// fp32 -> fp16 conversion
// ---------------------------------------------------------------------------
__global__ void cvt16(const float* __restrict__ in, _Float16* __restrict__ out, int n) {
  int i = blockIdx.x * 256 + threadIdx.x;
  if (i < n) out[i] = (_Float16)in[i];
}

// ---------------------------------------------------------------------------
// Setup 1: MGS QR of JF.T (96x64) with identity completion. Single wave.
// Writes Qr (64 cols x 96 rows, [k][n]), R (64x64), Qn (96x32, [n][t]).
// ---------------------------------------------------------------------------
__global__ void setup_qr(const float* __restrict__ A_eq, const float* __restrict__ A_ineq,
                         float* __restrict__ Qr_ws, float* __restrict__ R_ws,
                         float* __restrict__ Qn_ws) {
  __shared__ float Qc[96][97];   // column-major: Qc[col][row]
  __shared__ float R[64][65];
  const int lane = threadIdx.x;

  // Build columns of JF.T : col j = row j of JF = [[A_eq,0],[A_ineq,I]]
  for (int j = 0; j < 64; ++j)
    for (int r = lane; r < 96; r += 32) {
      float v;
      if (r < 64) v = (j < 32) ? A_eq[j * 64 + r] : A_ineq[(j - 32) * 64 + r];
      else        v = (j >= 32 && (r - 64) == (j - 32)) ? 1.0f : 0.0f;
      Qc[j][r] = v;
    }
  // completion columns: e_0..e_31
  for (int j = 64; j < 96; ++j)
    for (int r = lane; r < 96; r += 32)
      Qc[j][r] = (r == (j - 64)) ? 1.0f : 0.0f;
  for (int idx = lane; idx < 64 * 65; idx += 32) ((float*)R)[idx] = 0.0f;

  // Modified Gram-Schmidt with re-orthogonalization
  for (int j = 0; j < 96; ++j) {
    for (int pass = 0; pass < 2; ++pass) {
      for (int i = 0; i < j; ++i) {
        float part = 0.0f;
        for (int r = lane; r < 96; r += 32) part += Qc[i][r] * Qc[j][r];
        part = wred(part);
        for (int r = lane; r < 96; r += 32) Qc[j][r] -= part * Qc[i][r];
        if (i < 64 && j < 64 && lane == 0) R[i][j] += part;
      }
    }
    float part = 0.0f;
    for (int r = lane; r < 96; r += 32) part += Qc[j][r] * Qc[j][r];
    part = wred(part);
    float nrm = sqrtf(part);
    float inv = (nrm > 1e-20f) ? 1.0f / nrm : 0.0f;
    for (int r = lane; r < 96; r += 32) Qc[j][r] *= inv;
    if (j < 64 && lane == 0) R[j][j] = nrm;
  }

  for (int idx = lane; idx < 64 * 96; idx += 32) {
    int k = idx / 96, n = idx % 96;
    Qr_ws[idx] = Qc[k][n];
  }
  for (int idx = lane; idx < 64 * 64; idx += 32) {
    int i = idx / 64, k = idx % 64;
    R_ws[idx] = R[i][k];
  }
  for (int idx = lane; idx < 96 * 32; idx += 32) {
    int n = idx / 32, t = idx % 32;
    Qn_ws[idx] = Qc[64 + t][n];
  }
}

// ---------------------------------------------------------------------------
// Setup 2: Rinv = R^-1 (upper tri backsubst), Zmat[i][n] = sum_k Rinv[i][k]*Qr[n][k]
// zeta = bvec @ Zmat then solves min-norm JF zeta = bvec. Single wave.
// ---------------------------------------------------------------------------
__global__ void setup_zmat(const float* __restrict__ Qr_ws, const float* __restrict__ R_ws,
                           _Float16* __restrict__ Zmat16) {
  __shared__ float R[64][65];
  __shared__ float X[64][65];
  __shared__ float Qr[64][96];   // [k][n]
  const int lane = threadIdx.x;

  for (int idx = lane; idx < 64 * 64; idx += 32) R[idx / 64][idx % 64] = R_ws[idx];
  for (int idx = lane; idx < 64 * 96; idx += 32) Qr[idx / 96][idx % 96] = Qr_ws[idx];

  // back substitution: R * X = I, column c per lane
  for (int c = lane; c < 64; c += 32) {
    for (int i = 63; i >= 0; --i) {
      float s = (i == c) ? 1.0f : 0.0f;
      for (int k = i + 1; k < 64; ++k) s -= R[i][k] * X[k][c];
      X[i][c] = s / R[i][i];
    }
  }
  // Zmat[i][n] = sum_k X[i][k] * Qr[k][n]
  for (int idx = lane; idx < 64 * 96; idx += 32) {
    int i = idx / 96, n = idx % 96;
    float s = 0.0f;
    for (int k = 0; k < 64; ++k) s += X[i][k] * Qr[k][n];
    Zmat16[idx] = (_Float16)s;
  }
}

// ---------------------------------------------------------------------------
// WMMA GEMM: C(MxN) = act(A(MxK) @ B(KxN) + bias).  M % 128 == 0, K % 32 == 0.
// act: 0 = none, 1 = relu, 2 = LB + sigmoid * (UB-LB)
// ---------------------------------------------------------------------------
__global__ __launch_bounds__(256) void gemm_wmma(
    const _Float16* __restrict__ A, int lda,
    const _Float16* __restrict__ Bw, int ldb,
    const float* __restrict__ bias,
    float* __restrict__ Cf, _Float16* __restrict__ Ch, int ldc,
    int N, int K, int act) {
  __shared__ _Float16 As[128][40];   // 80B row stride: aligned + conflict-free
  __shared__ _Float16 Bs[32][72];    // 144B row stride

  const int tid  = threadIdx.x;
  const int lane = tid & 31;
  const int wv   = tid >> 5;
  const int wm   = wv & 3;          // wave M quadrant (0..3)
  const int wn   = wv >> 2;         // wave N half (0..1)
  const int row0 = blockIdx.y * 128;
  const int col0 = blockIdx.x * 64;

  const int am = tid >> 1;          // A tile row 0..127
  const int ak = (tid & 1) * 16;    // A tile col 0/16
  const int bk = tid >> 3;          // B tile row 0..31
  const int bn = (tid & 7) * 8;     // B tile col 0..56

  v8f acc[2][2];
#pragma unroll
  for (int i = 0; i < 2; ++i)
#pragma unroll
    for (int j = 0; j < 2; ++j) acc[i][j] = (v8f)0.0f;

  const int h   = lane >> 4;
  const int ml  = lane & 15;

  for (int k0 = 0; k0 < K; k0 += 32) {
    uint4 av = *(const uint4*)(A + (size_t)(row0 + am) * lda + k0 + ak);
    uint4 bv;
    if (col0 + bn < N) {
      bv = *(const uint4*)(Bw + (size_t)(k0 + bk) * ldb + col0 + bn);
    } else {
      bv.x = bv.y = bv.z = bv.w = 0u;
    }
    if (k0 + 32 < K)
      __builtin_prefetch(A + (size_t)(row0 + am) * lda + k0 + 32 + ak, 0, 1);
    __syncthreads();
    *(uint4*)&As[am][ak] = av;
    *(uint4*)&Bs[bk][bn] = bv;
    __syncthreads();

    // fragments
    v16h afr[2], bfr[2];
#pragma unroll
    for (int i = 0; i < 2; ++i) {
      const int m = wm * 32 + i * 16 + ml;
      union { unsigned u[8]; v16h v; } ua;
      const unsigned* rp = (const unsigned*)&As[m][0];
#pragma unroll
      for (int t = 0; t < 8; ++t)
        ua.u[t] = rp[(t >> 2) * 8 + h * 4 + (t & 3)];
      afr[i] = ua.v;
    }
#pragma unroll
    for (int j = 0; j < 2; ++j) {
      const int n0 = wn * 32 + j * 16;
      union { _Float16 e[16]; v16h v; } ub;
#pragma unroll
      for (int s = 0; s < 16; ++s)
        ub.e[s] = Bs[(s >> 3) * 16 + h * 8 + (s & 7)][n0 + ml];
      bfr[j] = ub.v;
    }
#pragma unroll
    for (int i = 0; i < 2; ++i)
#pragma unroll
      for (int j = 0; j < 2; ++j)
        acc[i][j] = __builtin_amdgcn_wmma_f32_16x16x32_f16(
            false, afr[i], false, bfr[j], (short)0, acc[i][j], false, false);
  }

  // epilogue
#pragma unroll
  for (int i = 0; i < 2; ++i) {
#pragma unroll
    for (int j = 0; j < 2; ++j) {
      const int col = col0 + wn * 32 + j * 16 + ml;
      if (col < N) {
        const float bb = bias ? bias[col] : 0.0f;
#pragma unroll
        for (int r = 0; r < 8; ++r) {
          const int row = row0 + wm * 32 + i * 16 + h * 8 + r;
          float v = acc[i][j][r] + bb;
          if (act == 1)      v = fmaxf(v, 0.0f);
          else if (act == 2) v = KLBC + (KUBC - KLBC) * (1.0f / (1.0f + __expf(-v)));
          if (Cf) Cf[(size_t)row * ldc + col] = v;
          if (Ch) Ch[(size_t)row * ldc + col] = (_Float16)v;
        }
      }
    }
  }
}

// ---------------------------------------------------------------------------
// Per-batch: M = Qn^T diag(p) Qn  (32x32 SPD), then in-place Gauss-Jordan
// inversion. One wave per batch element, 8 per block.
// ---------------------------------------------------------------------------
__global__ __launch_bounds__(256) void qtpq_inv(const float* __restrict__ p,
                                                const float* __restrict__ QnG,
                                                float* __restrict__ Minv) {
  __shared__ float Qn[96][32];
  __shared__ float Amat[8][32][33];
  __shared__ float ps[8][96];
  const int tid = threadIdx.x, lane = tid & 31, wv = tid >> 5;
  for (int i = tid; i < 96 * 32; i += 256) ((float*)Qn)[i] = QnG[i];
  __syncthreads();

  const int b = blockIdx.x * 8 + wv;
#pragma unroll
  for (int t = 0; t < 3; ++t) ps[wv][lane + 32 * t] = p[(size_t)b * 96 + lane + 32 * t];

  float m[32];
#pragma unroll
  for (int i = 0; i < 32; ++i) m[i] = 0.0f;
  for (int n = 0; n < 96; ++n) {
    const float wj = ps[wv][n] * Qn[n][lane];
#pragma unroll
    for (int i = 0; i < 32; ++i) m[i] += wj * Qn[n][i];
  }
#pragma unroll
  for (int i = 0; i < 32; ++i) Amat[wv][i][lane] = m[i];

  // in-place Gauss-Jordan (SPD, no pivoting); lane = column j
  for (int k = 0; k < 32; ++k) {
    const float pinv = 1.0f / Amat[wv][k][k];
    const float akj  = Amat[wv][k][lane];
    const float newk = (lane == k) ? pinv : akj * pinv;
    Amat[wv][k][lane] = newk;
    for (int i = 0; i < 32; ++i) {
      if (i == k) continue;
      const float f = Amat[wv][i][k];
      const float v = Amat[wv][i][lane];
      Amat[wv][i][lane] = (lane == k) ? (-f * pinv) : (v - f * newk);
    }
  }
  float* mo = Minv + (size_t)b * 1024;
#pragma unroll
  for (int i = 0; i < 32; ++i) mo[i * 32 + lane] = Amat[wv][i][lane];
}

// ---------------------------------------------------------------------------
// One Douglas-Rachford iteration (fused). One wave per batch row.
//   rhs   = p*(z-zeta) - [z_x + c ; 0]         (GAMMA/2 == 1)
//   rhs_n = rhs @ Qn ; w = Minv[b] @ rhs_n ; y = Qn @ w + zeta
//   v = 2y - z ; pg2 = [v_x, relu(v_s)] ; z' = z + pg2 - y
// ---------------------------------------------------------------------------
__global__ __launch_bounds__(256) void dr_iter(
    const float* __restrict__ zin, float* __restrict__ zout, float* __restrict__ yout,
    const float* __restrict__ p, const float* __restrict__ zeta,
    const float* __restrict__ parms, const float* __restrict__ Minv,
    const float* __restrict__ QnG) {
  __shared__ float Qn[96][32];
  __shared__ float scr[8][96];
  const int tid = threadIdx.x, lane = tid & 31, wv = tid >> 5;
  for (int i = tid; i < 96 * 32; i += 256) ((float*)Qn)[i] = QnG[i];
  __syncthreads();

  const int b = blockIdx.x * 8 + wv;
  const size_t rb = (size_t)b * 96;

  float zr[3], ztr[3], pr[3];
#pragma unroll
  for (int t = 0; t < 3; ++t) {
    const int i = lane + 32 * t;
    zr[t] = zin[rb + i]; ztr[t] = zeta[rb + i]; pr[t] = p[rb + i];
  }
  float cr[2];
  cr[0] = parms[(size_t)b * 128 + lane];
  cr[1] = parms[(size_t)b * 128 + 32 + lane];

#pragma unroll
  for (int t = 0; t < 3; ++t) {
    const int i = lane + 32 * t;
    const float g = (t < 2) ? (zr[t] + cr[t]) : 0.0f;   // i<64 iff t<2
    scr[wv][i] = pr[t] * (zr[t] - ztr[t]) - g;
  }
  // rhs_n[lane]
  float rn = 0.0f;
  for (int i = 0; i < 96; ++i) rn += scr[wv][i] * Qn[i][lane];
  scr[wv][lane] = rn;
  // w[lane] = Minv row dot rhs_n
  const float4* m4 = (const float4*)(Minv + (size_t)b * 1024 + lane * 32);
  float w = 0.0f;
#pragma unroll
  for (int t = 0; t < 8; ++t) {
    const float4 mv = m4[t];
    w += mv.x * scr[wv][4 * t] + mv.y * scr[wv][4 * t + 1] +
         mv.z * scr[wv][4 * t + 2] + mv.w * scr[wv][4 * t + 3];
  }
  scr[wv][32 + lane] = w;
  // y, prox_g2, z update
#pragma unroll
  for (int t = 0; t < 3; ++t) {
    const int i = lane + 32 * t;
    float y = ztr[t];
#pragma unroll
    for (int j = 0; j < 32; ++j) y += Qn[i][j] * scr[wv][32 + j];
    const float v  = 2.0f * y - zr[t];
    const float pg = (t < 2) ? v : fmaxf(v, 0.0f);
    zout[rb + i] = zr[t] + pg - y;
    yout[rb + i] = y;
  }
}

// ---------------------------------------------------------------------------
// Host orchestration
// ---------------------------------------------------------------------------
extern "C" void kernel_launch(void* const* d_in, const int* in_sizes, int n_in,
                              void* d_out, int out_size, void* d_ws, size_t ws_size,
                              hipStream_t stream) {
  const float* x      = (const float*)d_in[0];
  const float* parms  = (const float*)d_in[1];
  const float* A_eq   = (const float*)d_in[2];
  const float* A_ineq = (const float*)d_in[3];
  const float* W1a = (const float*)d_in[4];  const float* b1a = (const float*)d_in[5];
  const float* W2a = (const float*)d_in[6];  const float* b2a = (const float*)d_in[7];
  const float* W3a = (const float*)d_in[8];  const float* b3a = (const float*)d_in[9];
  const float* W1b = (const float*)d_in[10]; const float* b1b = (const float*)d_in[11];
  const float* W2b = (const float*)d_in[12]; const float* b2b = (const float*)d_in[13];
  const float* W3b = (const float*)d_in[14]; const float* b3b = (const float*)d_in[15];
  float* out = (float*)d_out;

  char* ws = (char*)d_ws;
  size_t off = 0;
  auto alloc = [&](size_t bytes) -> void* {
    void* ptr = (void*)(ws + off);
    off += (bytes + 255) & ~(size_t)255;
    return ptr;
  };
  _Float16* parms16 = (_Float16*)alloc((size_t)KB * KPARM * 2);
  _Float16* w1a16 = (_Float16*)alloc((size_t)KPARM * KHID * 2);
  _Float16* w2a16 = (_Float16*)alloc((size_t)KHID * KHID * 2);
  _Float16* w3a16 = (_Float16*)alloc((size_t)KHID * KN * 2);
  _Float16* w1b16 = (_Float16*)alloc((size_t)KPARM * KHID * 2);
  _Float16* w2b16 = (_Float16*)alloc((size_t)KHID * KHID * 2);
  _Float16* w3b16 = (_Float16*)alloc((size_t)KHID * KN * 2);
  _Float16* h1    = (_Float16*)alloc((size_t)KB * KHID * 2);
  _Float16* h2    = (_Float16*)alloc((size_t)KB * KHID * 2);
  float* p1   = (float*)alloc((size_t)KB * KN * 4);
  float* p2   = (float*)alloc((size_t)KB * KN * 4);
  float* zeta = (float*)alloc((size_t)KB * KN * 4);
  float* zbuf = (float*)alloc((size_t)KB * KN * 4);
  float* Qr_ws = (float*)alloc((size_t)KM * KN * 4);
  float* R_ws  = (float*)alloc((size_t)KM * KM * 4);
  float* Qn_ws = (float*)alloc((size_t)KN * 32 * 4);
  _Float16* Zmat16 = (_Float16*)alloc((size_t)KM * KN * 2);
  float* Minv = (float*)alloc((size_t)KB * 1024 * 4);
  (void)ws_size; (void)in_sizes; (void)n_in; (void)out_size;

  auto cvt = [&](const float* src, _Float16* dst, int n) {
    cvt16<<<(n + 255) / 256, 256, 0, stream>>>(src, dst, n);
  };
  cvt(parms, parms16, KB * KPARM);
  cvt(W1a, w1a16, KPARM * KHID); cvt(W2a, w2a16, KHID * KHID); cvt(W3a, w3a16, KHID * KN);
  cvt(W1b, w1b16, KPARM * KHID); cvt(W2b, w2b16, KHID * KHID); cvt(W3b, w3b16, KHID * KN);

  setup_qr<<<1, 32, 0, stream>>>(A_eq, A_ineq, Qr_ws, R_ws, Qn_ws);
  setup_zmat<<<1, 32, 0, stream>>>(Qr_ws, R_ws, Zmat16);

  const dim3 blk(256);
  const dim3 gHID((KHID + 63) / 64, KB / 128);
  const dim3 gN((KN + 63) / 64, KB / 128);

  // phase a MLP -> p1
  gemm_wmma<<<gHID, blk, 0, stream>>>(parms16, KPARM, w1a16, KHID, b1a, nullptr, h1, KHID, KHID, KPARM, 1);
  gemm_wmma<<<gHID, blk, 0, stream>>>(h1, KHID, w2a16, KHID, b2a, nullptr, h2, KHID, KHID, KHID, 1);
  gemm_wmma<<<gN,   blk, 0, stream>>>(h2, KHID, w3a16, KN,   b3a, p1, nullptr, KN, KN, KHID, 2);
  // phase b MLP -> p2
  gemm_wmma<<<gHID, blk, 0, stream>>>(parms16, KPARM, w1b16, KHID, b1b, nullptr, h1, KHID, KHID, KPARM, 1);
  gemm_wmma<<<gHID, blk, 0, stream>>>(h1, KHID, w2b16, KHID, b2b, nullptr, h2, KHID, KHID, KHID, 1);
  gemm_wmma<<<gN,   blk, 0, stream>>>(h2, KHID, w3b16, KN,   b3b, p2, nullptr, KN, KN, KHID, 2);
  // zeta = parms[:,64:128] @ Zmat  (iteration-invariant min-norm solution)
  gemm_wmma<<<gN, blk, 0, stream>>>(parms16 + 64, KPARM, Zmat16, KN, nullptr, zeta, nullptr, KN, KN, KM, 0);

  const int nb = KB / 8;
  // phase a: 1 DR step
  qtpq_inv<<<nb, blk, 0, stream>>>(p1, Qn_ws, Minv);
  dr_iter<<<nb, blk, 0, stream>>>(x, zbuf, out, p1, zeta, parms, Minv, Qn_ws);
  // phase b: 3 DR steps
  qtpq_inv<<<nb, blk, 0, stream>>>(p2, Qn_ws, Minv);
  for (int s = 0; s < 3; ++s)
    dr_iter<<<nb, blk, 0, stream>>>(zbuf, zbuf, out, p2, zeta, parms, Minv, Qn_ws);
}
